// Encoder_71150428225961
// MI455X (gfx1250) — matrix operand order repaired and checked by reading, not compiled, not run on hardware
//
#include <hip/hip_runtime.h>

// ---------------------------------------------------------------------------
// Problem dims (compile-time)
// ---------------------------------------------------------------------------
constexpr int B_  = 32;
constexpr int S_  = 256;
constexpr int D_  = 1024;
constexpr int H_  = 16;
constexpr int DH_ = 64;
constexpr int HID_= 4096;
constexpr int NTOK = B_ * S_;            // 8192 rows

// ---------------------------------------------------------------------------
// CDNA5 async global->LDS path. Probe rounds 2-3 established the builtin's
// signature: (native int-vec16B ptr [global], native int-vec16B ptr [LDS],
// imm offset, imm cpol). HIP's int4 is a struct -> must use a native vector.
// ---------------------------------------------------------------------------
#define LDS_AS __attribute__((address_space(3)))
#define GLB_AS __attribute__((address_space(1)))
typedef int ivec4 __attribute__((vector_size(16)));   // native vector, 16B
#if defined(__AMDGCN__) && \
    __has_builtin(__builtin_amdgcn_global_load_async_to_lds_b128) && \
    __has_builtin(__builtin_amdgcn_s_wait_asynccnt)
#define HAVE_GLDS_ASYNC 1
#else
#define HAVE_GLDS_ASYNC 0
#endif

// ---------------------------------------------------------------------------
// WMMA types (CDNA5 / gfx1250, wave32)
// ---------------------------------------------------------------------------
typedef __attribute__((ext_vector_type(16))) __bf16 v16bf;
typedef __attribute__((ext_vector_type(8)))  float  v8f;

union FragBF {                 // 16 bf16 elements = 8 VGPRs, A or B operand
    v16bf bf;
    int4  i4[2];
    unsigned short u16[16];
};

__device__ __forceinline__ v8f wmma_bf16(const FragBF& a, const FragBF& b, v8f c) {
    // (neg_a, A, neg_b, B, c_mod, C, reuse_a, reuse_b)
    return __builtin_amdgcn_wmma_f32_16x16x32_bf16(false, a.bf, false, b.bf,
                                                   (short)0, c, false, false);
}

__device__ __forceinline__ unsigned short f2bf(float f) {
    unsigned int u = __float_as_uint(f);
    unsigned int r = u + 0x7FFFu + ((u >> 16) & 1u);   // round-to-nearest-even
    return (unsigned short)(r >> 16);
}

// ---------------------------------------------------------------------------
// fp32 [R][C] -> bf16 transposed [C][R] (weights; tiled 32x32 via LDS)
// grid = (C/32, R/32), block = 256 (32x8)
// ---------------------------------------------------------------------------
__global__ __launch_bounds__(256) void cvtT_bf16_kernel(const float* __restrict__ in,
                                                        unsigned short* __restrict__ out,
                                                        int R, int C) {
    __shared__ float tile[32][33];
    const int c0 = blockIdx.x * 32;
    const int r0 = blockIdx.y * 32;
    const int tx = threadIdx.x & 31, ty = threadIdx.x >> 5;
#pragma unroll
    for (int r = 0; r < 32; r += 8)
        tile[r + ty][tx] = in[(long)(r0 + r + ty) * C + c0 + tx];
    __syncthreads();
#pragma unroll
    for (int r = 0; r < 32; r += 8)
        out[(long)(c0 + r + ty) * R + r0 + tx] = f2bf(tile[tx][r + ty]);
}

// ---------------------------------------------------------------------------
// Embedding * sqrt(D) + sinusoidal positional encoding  ->  x (fp32)
// ---------------------------------------------------------------------------
__global__ __launch_bounds__(256) void embed_pe_kernel(const int* __restrict__ src,
                                                       const float* __restrict__ emb,
                                                       float* __restrict__ x) {
    long i  = (long)blockIdx.x * 256 + threadIdx.x;      // covers B*S*D exactly
    int  d  = (int)(i & (D_ - 1));
    long bs = i >> 10;                                   // /D_
    int  s  = (int)(bs & (S_ - 1));
    int  tok = src[bs];
    float f   = __expf(-(float)(d & ~1) * (9.210340371976184f / (float)D_));
    float arg = (float)s * f;
    float pe  = (d & 1) ? __cosf(arg) : __sinf(arg);
    x[i] = emb[(long)tok * D_ + d] * 32.0f + pe;         // sqrt(1024)=32
}

// ---------------------------------------------------------------------------
// LayerNorm over D=1024, one workgroup (256 thr = 8 waves) per row
// ---------------------------------------------------------------------------
template<bool OUTBF>
__global__ __launch_bounds__(256) void ln_kernel(const float* __restrict__ x,
                                                 const float* __restrict__ gamma,
                                                 const float* __restrict__ beta,
                                                 unsigned short* __restrict__ outb,
                                                 float* __restrict__ outf) {
    const int row = blockIdx.x;
    const float* xr = x + (long)row * D_;
    const int tid = threadIdx.x;
    float v[4], s = 0.f, s2 = 0.f;
#pragma unroll
    for (int i = 0; i < 4; ++i) {
        v[i] = xr[tid + i * 256];
        s += v[i]; s2 += v[i] * v[i];
    }
#pragma unroll
    for (int m = 1; m < 32; m <<= 1) { s += __shfl_xor(s, m); s2 += __shfl_xor(s2, m); }
    __shared__ float ls[8], ls2[8];
    if ((tid & 31) == 0) { ls[tid >> 5] = s; ls2[tid >> 5] = s2; }
    __syncthreads();
    s = 0.f; s2 = 0.f;
#pragma unroll
    for (int w = 0; w < 8; ++w) { s += ls[w]; s2 += ls2[w]; }
    float mean = s * (1.0f / D_);
    float var  = s2 * (1.0f / D_) - mean * mean;
    float rstd = rsqrtf(var + 1e-5f);
#pragma unroll
    for (int i = 0; i < 4; ++i) {
        int c = tid + i * 256;
        float y = (v[i] - mean) * rstd * gamma[c] + beta[c];
        if (OUTBF) outb[(long)row * D_ + c] = f2bf(y);
        else       outf[(long)row * D_ + c] = y;
    }
}

// ---------------------------------------------------------------------------
// bf16 WMMA GEMM:  out[M,N] = A[M,K] @ WT[N,K]^T + bias  (+epilogue)
//   EPI 0: store bf16 | EPI 1: exact GELU, bf16 | EPI 2: += resid, fp32
// Block tile 128x128, BK=64, 8 waves (4x2), wave tile 32x64 -> 16 WMMAs/stage.
// Both LDS tiles are [row][64] with row stride padded to 72 halves
// (gcd(36,64)=4 -> 16 lanes spread over all banks, 16B alignment kept).
// Async path: global_load_async_to_lds_b128 double-buffer, s_wait_asynccnt.
// ---------------------------------------------------------------------------
constexpr int LSTR = 72;                 // padded LDS row stride (halves)

__device__ __forceinline__ void gemm_compute_stage(
    const unsigned short* lA, const unsigned short* lB,
    int wm, int wn, int ln16, int lh, v8f (&acc)[2][4]) {
#pragma unroll
    for (int ksub = 0; ksub < 2; ++ksub) {
        FragBF a[2], b[4];
#pragma unroll
        for (int i = 0; i < 2; ++i) {
            const int4* p = (const int4*)(lA + (wm * 32 + i * 16 + ln16) * LSTR +
                                          ksub * 32 + lh * 16);
            a[i].i4[0] = p[0];
            a[i].i4[1] = p[1];           // +8 halves
        }
#pragma unroll
        for (int j = 0; j < 4; ++j) {
            const int4* p = (const int4*)(lB + (wn * 64 + j * 16 + ln16) * LSTR +
                                          ksub * 32 + lh * 16);
            b[j].i4[0] = p[0];
            b[j].i4[1] = p[1];
        }
#pragma unroll
        for (int i = 0; i < 2; ++i)
#pragma unroll
            for (int j = 0; j < 4; ++j)
                acc[i][j] = wmma_bf16(a[i], b[j], acc[i][j]);
    }
}

#if HAVE_GLDS_ASYNC
__device__ __forceinline__ void gemm_issue_async(
    const unsigned short* A, const unsigned short* WT,
    unsigned short* lA, unsigned short* lB,
    long row0, long col0, int K, int k0, int tid) {
#pragma unroll
    for (int t = 0; t < 4; ++t) {
        int idx = tid + t * 256;
        int r = idx >> 3, c8 = (idx & 7) * 8;
        __builtin_amdgcn_global_load_async_to_lds_b128(
            (GLB_AS ivec4*)(A + (row0 + r) * K + k0 + c8),
            (LDS_AS ivec4*)(lA + r * LSTR + c8), 0, 0);
        __builtin_amdgcn_global_load_async_to_lds_b128(
            (GLB_AS ivec4*)(WT + (col0 + r) * K + k0 + c8),
            (LDS_AS ivec4*)(lB + r * LSTR + c8), 0, 0);
    }
}
#endif

template<int EPI>
__global__ __launch_bounds__(256) void gemm_bf16_kernel(
    const unsigned short* __restrict__ A, const unsigned short* __restrict__ WT,
    const float* __restrict__ bias, const float* __restrict__ resid,
    unsigned short* __restrict__ outb, float* __restrict__ outf,
    int M, int N, int K) {
    const int tid  = threadIdx.x;
    const int lane = tid & 31, wave = tid >> 5;
    const int wm = wave >> 1, wn = wave & 1;
    const int ln16 = lane & 15, lh = lane >> 4;
    const long row0 = (long)blockIdx.y * 128;
    const long col0 = (long)blockIdx.x * 128;

    v8f acc[2][4];
    {
        v8f z = {};
#pragma unroll
        for (int i = 0; i < 2; ++i)
#pragma unroll
            for (int j = 0; j < 4; ++j) acc[i][j] = z;
    }

    const int nst = K >> 6;              // K is a multiple of 64 (>=1024)

#if HAVE_GLDS_ASYNC
    __shared__ __align__(16) unsigned short lA[2][128 * LSTR];
    __shared__ __align__(16) unsigned short lB[2][128 * LSTR];
    gemm_issue_async(A, WT, lA[0], lB[0], row0, col0, K, 0, tid);
    gemm_issue_async(A, WT, lA[1], lB[1], row0, col0, K, 64, tid);
    for (int s = 0; s < nst - 1; ++s) {
        __builtin_amdgcn_s_wait_asynccnt(8);   // stage s landed (s+1 in flight)
        __syncthreads();
        gemm_compute_stage(lA[s & 1], lB[s & 1], wm, wn, ln16, lh, acc);
        __syncthreads();                       // all waves done reading buf s&1
        if (s + 2 < nst)
            gemm_issue_async(A, WT, lA[s & 1], lB[s & 1], row0, col0, K,
                             (s + 2) * 64, tid);
    }
    __builtin_amdgcn_s_wait_asynccnt(0);
    __syncthreads();
    gemm_compute_stage(lA[(nst - 1) & 1], lB[(nst - 1) & 1], wm, wn, ln16, lh, acc);
#else
    __shared__ __align__(16) unsigned short lA1[128 * LSTR];
    __shared__ __align__(16) unsigned short lB1[128 * LSTR];
    for (int s = 0; s < nst; ++s) {
        const int k0 = s * 64;
        if (s) __syncthreads();
#pragma unroll
        for (int t = 0; t < 4; ++t) {
            int idx = tid + t * 256;
            int r = idx >> 3, c8 = (idx & 7) * 8;
            *(int4*)(lA1 + r * LSTR + c8) =
                *(const int4*)(A + (row0 + r) * K + k0 + c8);
            *(int4*)(lB1 + r * LSTR + c8) =
                *(const int4*)(WT + (col0 + r) * K + k0 + c8);
        }
        if (k0 + 64 < K) {                     // gfx1250 global_prefetch_b8
            __builtin_prefetch(A + (row0 + (tid >> 3)) * K + k0 + 64, 0, 1);
            __builtin_prefetch(WT + (col0 + (tid >> 3)) * K + k0 + 64, 0, 1);
        }
        __syncthreads();
        gemm_compute_stage(lA1, lB1, wm, wn, ln16, lh, acc);
    }
#endif

    // ---- epilogue (C layout: reg r -> row r + 8*(lane>=16), col = lane&15) ----
#pragma unroll
    for (int i = 0; i < 2; ++i) {
#pragma unroll
        for (int j = 0; j < 4; ++j) {
            long col = col0 + wn * 64 + j * 16 + ln16;
            float bv = bias[col];
#pragma unroll
            for (int r = 0; r < 8; ++r) {
                long row = row0 + wm * 32 + i * 16 + r + lh * 8;
                float v = acc[i][j][r] + bv;
                if (EPI == 1) v = 0.5f * v * (1.0f + erff(v * 0.7071067811865476f));
                if (EPI == 2) outf[row * N + col] = v + resid[row * N + col];
                else          outb[row * N + col] = f2bf(v);
            }
        }
    }
}

// ---------------------------------------------------------------------------
// Flash attention, bf16 WMMA, key-padding mask.
// grid = B*H*2 (2 query chunks of 128); block = 256 thr = 8 waves.
// Padded LDS strides: lK rows 72 halves, lVT rows 136, lP rows 40.
// ---------------------------------------------------------------------------
constexpr int KSTR = 72;
constexpr int VSTR = 136;
constexpr int PSTR = 40;

__global__ __launch_bounds__(256) void attn_kernel(
    const unsigned short* __restrict__ q, const unsigned short* __restrict__ kv,
    const int* __restrict__ lens, unsigned short* __restrict__ o) {
    __shared__ __align__(16) unsigned short lK [128 * KSTR];       // [key][dh]
    __shared__ __align__(16) unsigned short lVT[DH_ * VSTR];       // [dh][key]
    __shared__ __align__(16) unsigned short lP [8 * 16 * PSTR];    // per-wave P
    const int blk = blockIdx.x;
    const int qc  = blk & 1;
    const int bh  = blk >> 1;
    const int h   = bh % H_;
    const int b   = bh / H_;
    const int tid = threadIdx.x, lane = tid & 31, wave = tid >> 5;
    const int ln16 = lane & 15, lh = lane >> 4;
    const int q0 = qc * 128 + wave * 16;
    const int len = lens[b];
    const float scale = 0.125f;                  // 1/sqrt(DH)

    FragBF qa[2];                                // Q fragments held in registers
    {
        const unsigned short* qp = q + ((long)(b * S_ + q0 + ln16) * H_ + h) * DH_;
#pragma unroll
        for (int ks = 0; ks < 2; ++ks) {
            int kb = ks * 32 + lh * 16;
            qa[ks].i4[0] = *(const int4*)(qp + kb);
            qa[ks].i4[1] = *(const int4*)(qp + kb + 8);
        }
    }

    v8f oacc[4];
    { v8f z = {}; for (int t = 0; t < 4; ++t) oacc[t] = z; }
    float mrun[8], lrun[8];
#pragma unroll
    for (int i = 0; i < 8; ++i) { mrun[i] = -1e30f; lrun[i] = 0.f; }

    for (int kc = 0; kc < 2; ++kc) {
        __syncthreads();
        // stage K chunk + transposed V chunk
#pragma unroll
        for (int p = 0; p < 4; ++p) {
            int idx = tid + p * 256;
            int r   = idx >> 3;                 // key 0..127
            int c8  = (idx & 7) * 8;            // dh chunk
            long gk = ((long)(b * S_ + kc * 128 + r) * (2 * D_)) + h * DH_ + c8;
            *(int4*)(lK + r * KSTR + c8) = *(const int4*)(kv + gk);
            unsigned short tmp[8];
            *(int4*)tmp = *(const int4*)(kv + gk + D_);
#pragma unroll
            for (int j = 0; j < 8; ++j) lVT[(c8 + j) * VSTR + r] = tmp[j];
        }
        __syncthreads();

        for (int ks = 0; ks < 4; ++ks) {        // 32 keys per step
            // ---- S = Q @ K^T (two 16x16 tiles) ----
            v8f st[2];
#pragma unroll
            for (int t = 0; t < 2; ++t) {
                int keyloc = ks * 32 + t * 16 + ln16;
                FragBF kb0, kb1;
                const int4* p0 = (const int4*)(lK + keyloc * KSTR + lh * 16);
                kb0.i4[0] = p0[0]; kb0.i4[1] = p0[1];
                const int4* p1 = (const int4*)(lK + keyloc * KSTR + 32 + lh * 16);
                kb1.i4[0] = p1[0]; kb1.i4[1] = p1[1];
                v8f c = {};
                c = wmma_bf16(qa[0], kb0, c);
                c = wmma_bf16(qa[1], kb1, c);
                st[t] = c;
            }
            // ---- mask + online softmax ----
            int key0 = kc * 128 + ks * 32 + ln16;
            bool ok0 = key0 < len;
            bool ok1 = (key0 + 16) < len;
#pragma unroll
            for (int i = 0; i < 8; ++i) {
                float a0 = ok0 ? st[0][i] * scale : -1e30f;
                float a1 = ok1 ? st[1][i] * scale : -1e30f;
                float t = fmaxf(a0, a1);
                t = fmaxf(t, __shfl_xor(t, 1));
                t = fmaxf(t, __shfl_xor(t, 2));
                t = fmaxf(t, __shfl_xor(t, 4));
                t = fmaxf(t, __shfl_xor(t, 8));
                float mn   = fmaxf(mrun[i], t);
                float corr = __expf(mrun[i] - mn);
                mrun[i] = mn;
                float p0 = __expf(a0 - mn);
                float p1 = __expf(a1 - mn);
                float rs = p0 + p1;
                rs += __shfl_xor(rs, 1);
                rs += __shfl_xor(rs, 2);
                rs += __shfl_xor(rs, 4);
                rs += __shfl_xor(rs, 8);
                lrun[i] = lrun[i] * corr + rs;
#pragma unroll
                for (int t4 = 0; t4 < 4; ++t4) oacc[t4][i] *= corr;
                int m = i + lh * 8;
                unsigned short* pp = lP + wave * (16 * PSTR) + m * PSTR;
                pp[ln16]      = f2bf(p0);
                pp[16 + ln16] = f2bf(p1);
            }
            __builtin_amdgcn_wave_barrier();    // keep DS writes before reads
            // ---- P @ V ----
            FragBF pf;
            {
                const int4* pp = (const int4*)(lP + wave * (16 * PSTR) +
                                               ln16 * PSTR + lh * 16);
                pf.i4[0] = pp[0];
                pf.i4[1] = pp[1];
            }
#pragma unroll
            for (int nt = 0; nt < 4; ++nt) {
                FragBF vb;
                const int4* vp =
                    (const int4*)(lVT + (nt * 16 + ln16) * VSTR + ks * 32 + lh * 16);
                vb.i4[0] = vp[0]; vb.i4[1] = vp[1];
                oacc[nt] = wmma_bf16(pf, vb, oacc[nt]);
            }
        }
    }
    // ---- normalize and store o[b, q, h*DH + dh] ----
#pragma unroll
    for (int i = 0; i < 8; ++i) {
        float inv = 1.0f / lrun[i];
        int m = i + lh * 8;
        unsigned short* op = o + ((long)(b * S_ + q0 + m) * H_ + h) * DH_;
#pragma unroll
        for (int nt = 0; nt < 4; ++nt) op[nt * 16 + ln16] = f2bf(oacc[nt][i] * inv);
    }
}

// ---------------------------------------------------------------------------
// Host orchestration
// ---------------------------------------------------------------------------
extern "C" void kernel_launch(void* const* d_in, const int* in_sizes, int n_in,
                              void* d_out, int out_size, void* d_ws, size_t ws_size,
                              hipStream_t stream) {
    const int*   src  = (const int*)  d_in[0];
    const int*   slen = (const int*)  d_in[1];
    const float* emb  = (const float*)d_in[2];
    const float* Wq   = (const float*)d_in[3];
    const float* bq   = (const float*)d_in[4];
    const float* Wkv  = (const float*)d_in[5];
    const float* bkv  = (const float*)d_in[6];
    const float* Wo   = (const float*)d_in[7];
    const float* bo   = (const float*)d_in[8];
    const float* lnas = (const float*)d_in[9];
    const float* lnab = (const float*)d_in[10];
    const float* W1   = (const float*)d_in[11];
    const float* b1   = (const float*)d_in[12];
    const float* W2   = (const float*)d_in[13];
    const float* b2   = (const float*)d_in[14];
    const float* lnfs = (const float*)d_in[15];
    const float* lnfb = (const float*)d_in[16];
    const float* lnos = (const float*)d_in[17];
    const float* lnob = (const float*)d_in[18];

    char* ws = (char*)d_ws;
    size_t off = 0;
    auto carve = [&](size_t bytes) -> char* {
        char* p = ws + off;
        off += (bytes + 255) & ~(size_t)255;
        return p;
    };
    float*          x    = (float*)         carve((size_t)NTOK * D_  * 4);
    unsigned short* xn   = (unsigned short*)carve((size_t)NTOK * D_  * 2);
    unsigned short* qb   = (unsigned short*)carve((size_t)NTOK * D_  * 2);
    unsigned short* kvb  = (unsigned short*)carve((size_t)NTOK * 2*D_* 2);
    unsigned short* ob   = (unsigned short*)carve((size_t)NTOK * D_  * 2);
    unsigned short* hb   = (unsigned short*)carve((size_t)NTOK * HID_* 2);
    unsigned short* WqT  = (unsigned short*)carve((size_t)D_ * D_   * 2); // [N][K]
    unsigned short* WkvT = (unsigned short*)carve((size_t)D_ * 2*D_ * 2);
    unsigned short* WoT  = (unsigned short*)carve((size_t)D_ * D_   * 2);
    unsigned short* W1T  = (unsigned short*)carve((size_t)D_ * HID_ * 2);
    unsigned short* W2T  = (unsigned short*)carve((size_t)HID_ * D_ * 2);

    auto cvtT = [&](const float* in, unsigned short* out, int R, int C) {
        cvtT_bf16_kernel<<<dim3(C / 32, R / 32), 256, 0, stream>>>(in, out, R, C);
    };
    cvtT(Wq,  WqT,  D_,  D_);       // [D][D]    -> [D][D]
    cvtT(Wkv, WkvT, D_,  2 * D_);   // [D][2D]   -> [2D][D]
    cvtT(Wo,  WoT,  D_,  D_);
    cvtT(W1,  W1T,  D_,  HID_);     // [D][HID]  -> [HID][D]
    cvtT(W2,  W2T,  HID_, D_);      // [HID][D]  -> [D][HID]

    embed_pe_kernel<<<(NTOK * D_) / 256, 256, 0, stream>>>(src, emb, x);

    for (int l = 0; l < 6; ++l) {
        // --- attention block ---
        ln_kernel<true><<<NTOK, 256, 0, stream>>>(x, lnas, lnab, xn, nullptr);
        gemm_bf16_kernel<0><<<dim3(D_ / 128, NTOK / 128), 256, 0, stream>>>(
            xn, WqT, bq, nullptr, qb, nullptr, NTOK, D_, D_);
        gemm_bf16_kernel<0><<<dim3(2 * D_ / 128, NTOK / 128), 256, 0, stream>>>(
            xn, WkvT, bkv, nullptr, kvb, nullptr, NTOK, 2 * D_, D_);
        attn_kernel<<<B_ * H_ * 2, 256, 0, stream>>>(qb, kvb, slen, ob);
        gemm_bf16_kernel<2><<<dim3(D_ / 128, NTOK / 128), 256, 0, stream>>>(
            ob, WoT, bo, x, nullptr, x, NTOK, D_, D_);
        // --- FFN block ---
        ln_kernel<true><<<NTOK, 256, 0, stream>>>(x, lnfs, lnfb, xn, nullptr);
        gemm_bf16_kernel<1><<<dim3(HID_ / 128, NTOK / 128), 256, 0, stream>>>(
            xn, W1T, b1, nullptr, hb, nullptr, NTOK, HID_, D_);
        gemm_bf16_kernel<2><<<dim3(D_ / 128, NTOK / 128), 256, 0, stream>>>(
            hb, W2T, b2, x, nullptr, x, NTOK, D_, HID_);
    }
    ln_kernel<false><<<NTOK, 256, 0, stream>>>(x, lnos, lnob, nullptr, (float*)d_out);
}
